// DynamicAgg_38319698214924
// MI455X (gfx1250) — matrix operand order repaired and checked by reading, not compiled, not run on hardware
//
#include <hip/hip_runtime.h>
#include <hip/hip_bf16.h>

// ---------------------------------------------------------------------------
// DynamicAgg on MI455X (gfx1250, wave32, WMMA f32_16x16x32_f16)
//
// Algebraic rewrite: new_adj (b,c,m,n) is only contracted with pc over n, so
//   pt[b,m,c,t] = ft[b,m,c,:] @ G[b,c],  G[b,c] = fc[b,:,c,:]^T @ pc[b,:,c,:]
// G is 24x24 per (b,c) -> the 268MB adjacency never touches HBM.
// ---------------------------------------------------------------------------

typedef __attribute__((ext_vector_type(16))) _Float16 v16h;
typedef __attribute__((ext_vector_type(8)))  _Float16 v8h;
typedef __attribute__((ext_vector_type(8)))  float    v8f;

static constexpr int Bb = 4, Mm = 512, Nn = 512, Tt = 24, CHN = 64;
#define LN_EPS 1e-5f

__device__ __forceinline__ int lane_id() { return (int)(threadIdx.x & 31); }

// A fragment (16xK, 16-bit): 16x32 tile, contiguous k order (same as load_frag_b).
// lane<16: M=lane, K(e)= kb + 16*hi + e; rows stored row-major in LDS.
// -> two contiguous 16B reads per lane.
__device__ __forceinline__ v16h load_frag_a(const _Float16* base, int stride,
                                            int m0, int kb) {
  int l  = lane_id();
  int m  = m0 + (l & 15);
  int hi = l >> 4;
  const _Float16* p = base + m * stride + kb + 16 * hi;
  union { v16h v; v8h h[2]; } u;
  u.h[0] = *(const v8h*)(p);
  u.h[1] = *(const v8h*)(p + 8);
  return u.v;
}

// B fragment (Kx16, 16-bit): storage is row-major [N][K] (each output column's
// K values contiguous) -> single contiguous 32B read per lane.
// lane<16: N=lane, K(e) = kb + 16*hi + e.
__device__ __forceinline__ v16h load_frag_b(const _Float16* base, int stride,
                                            int n0, int kb) {
  int l  = lane_id();
  int n  = n0 + (l & 15);
  int hi = l >> 4;
  const _Float16* p = base + n * stride + kb + 16 * hi;
  return *(const v16h*)p;
}

__device__ __forceinline__ v8f wmma16(v16h a, v16h b, v8f c) {
  return __builtin_amdgcn_wmma_f32_16x16x32_f16(
      /*neg_a=*/false, a, /*neg_b=*/false, b,
      /*c_mod=*/(short)0, c, /*reuse_a=*/false, /*reuse_b=*/false);
}

// ---------------------------------------------------------------------------
// Fused 4-layer pointwise MLP, cin<=32 (7 or 1), hidden/out=64, ReLU between.
// Input f32 (row, cin, t) point-layout; output f16 (row, 64, t).
// 128 points per block; each of the 8 waves owns 16 point rows -> no barriers
// between layers.
// ---------------------------------------------------------------------------
__global__ void __launch_bounds__(256)
mlp4_small_kernel(const float* __restrict__ Xin, _Float16* __restrict__ Yout,
                  const float* __restrict__ W0, const float* __restrict__ Bi0,
                  const float* __restrict__ W1, const float* __restrict__ Bi1,
                  const float* __restrict__ W2, const float* __restrict__ Bi2,
                  const float* __restrict__ W3, const float* __restrict__ Bi3,
                  int cin, int P) {
  __shared__ __align__(128) _Float16 Xs[128 * 72];     // 18 KB, stride 72 -> no bank conflicts
  __shared__ __align__(128) _Float16 Ws[4 * 64 * 64];  // 32 KB, [layer][o][k]
  __shared__ float Bs[4 * 64];

  const int tid = (int)threadIdx.x;

  // zero-fill weights (K padding for layer 0), then stage f32 -> f16
  for (int i = tid; i < 4 * 64 * 64; i += 256) Ws[i] = (_Float16)0.f;
  __syncthreads();
  for (int i = tid; i < 64 * cin; i += 256)
    Ws[(i / cin) * 64 + (i % cin)] = (_Float16)W0[i];
  for (int i = tid; i < 64 * 64; i += 256) {
    Ws[4096 + i]  = (_Float16)W1[i];
    Ws[8192 + i]  = (_Float16)W2[i];
    Ws[12288 + i] = (_Float16)W3[i];
  }
  if (tid < 64) {
    Bs[tid]       = Bi0[tid];
    Bs[64 + tid]  = Bi1[tid];
    Bs[128 + tid] = Bi2[tid];
    Bs[192 + tid] = Bi3[tid];
  }

  // stage input tile: point p=(row,t), t fastest; zero-pad channels >= cin
  const int p0 = (int)blockIdx.x * 128;
  for (int i = tid; i < 128 * 64; i += 256) {
    int j = i >> 6, cc = i & 63;
    int p = p0 + j;
    float v = 0.f;
    if (p < P && cc < cin) {
      int row = p / Tt, t = p - row * Tt;
      v = Xin[((size_t)row * cin + cc) * Tt + t];
    }
    Xs[j * 72 + cc] = (_Float16)v;
  }
  __syncthreads();

  const int w = tid >> 5, m0 = w * 16;
  const int l = lane_id(), col = l & 15, hi = l >> 4;

#pragma unroll
  for (int L = 0; L < 4; L++) {
    const _Float16* Wl = Ws + L * 4096;
    v16h a0 = load_frag_a(Xs, 72, m0, 0);
    v16h a1 = a0;
    if (L > 0) a1 = load_frag_a(Xs, 72, m0, 32);

    v8f acc[4];
#pragma unroll
    for (int nt = 0; nt < 4; nt++) {
      v8f z = {0.f, 0.f, 0.f, 0.f, 0.f, 0.f, 0.f, 0.f};
      z = wmma16(a0, load_frag_b(Wl, 64, nt * 16, 0), z);
      if (L > 0) z = wmma16(a1, load_frag_b(Wl, 64, nt * 16, 32), z);
      acc[nt] = z;
    }

    if (L < 3) {
      // bias + ReLU, write back to this wave's own rows as f16
#pragma unroll
      for (int nt = 0; nt < 4; nt++) {
        int o = nt * 16 + col;
        float bia = Bs[L * 64 + o];
#pragma unroll
        for (int r = 0; r < 8; r++) {
          int m = m0 + r + 8 * hi;
          float v = acc[nt][r] + bia;
          Xs[m * 72 + o] = (_Float16)(v > 0.f ? v : 0.f);
        }
      }
    } else {
      // final layer: bias only, store f16 to (row, c, t) layout
#pragma unroll
      for (int nt = 0; nt < 4; nt++) {
        int o = nt * 16 + col;
        float bia = Bs[192 + o];
#pragma unroll
        for (int r = 0; r < 8; r++) {
          int m = m0 + r + 8 * hi;
          int p = p0 + m;
          if (p < P) {
            int row = p / Tt, t = p - row * Tt;
            Yout[((size_t)row * CHN + o) * Tt + t] = (_Float16)(acc[nt][r] + bia);
          }
        }
      }
    }
  }
}

// ---------------------------------------------------------------------------
// Per-(b,c): G = fc^T @ pc (24x24 over K=N=512), then pt = ft @ G.
// One block per (b,c) = 256 blocks. fc/pc staged t-major (transposed) in two
// 256-n chunks so static LDS stays under 64 KB; ft staging reuses the buffer.
// pt written point-major [point][64] f32 for contiguous LayerNorm staging.
// ---------------------------------------------------------------------------
__global__ void __launch_bounds__(256)
adj_apply_kernel(const _Float16* __restrict__ FT, const _Float16* __restrict__ FC,
                 const _Float16* __restrict__ PC, float* __restrict__ PTpm) {
  constexpr int NCH = 256;                             // n-chunk size
  __shared__ __align__(128) _Float16 S[32 * NCH * 2];  // 32 KB: fcT | pcT (reused for ftS)
  __shared__ __align__(128) _Float16 GB[32 * 32];      // G transposed, [t][t'] zero-padded

  const int tid = (int)threadIdx.x;
  const int b = (int)blockIdx.x >> 6, c = (int)blockIdx.x & 63;
  _Float16* fcT = S;
  _Float16* pcT = S + 32 * NCH;

  for (int i = tid; i < 32 * 32; i += 256) GB[i] = (_Float16)0.f;

  const int w = tid >> 5, l = lane_id(), col = l & 15, hi = l >> 4;
  const int Mt = (w >> 1) & 1, Nt = w & 1;  // gram tile for waves 0..3

  v8f gacc = {0.f, 0.f, 0.f, 0.f, 0.f, 0.f, 0.f, 0.f};
  for (int chunk = 0; chunk < 2; chunk++) {
    __syncthreads();  // previous chunk fully consumed before overwrite
    for (int i = tid; i < NCH * Tt; i += 256) {
      int nl = i / Tt, t = i - nl * Tt;
      int n = chunk * NCH + nl;
      size_t g = (((size_t)(b * Nn + n)) * CHN + c) * Tt + t;
      fcT[t * NCH + nl] = FC[g];   // A: rows = t1, K = n (contiguous)
      pcT[t * NCH + nl] = PC[g];   // B: rows = t2, K = n (contiguous)
    }
    __syncthreads();
    if (w < 4) {
#pragma unroll
      for (int ks = 0; ks < 8; ks++) {
        v16h a  = load_frag_a(fcT, NCH, Mt * 16, ks * 32);
        v16h bb = load_frag_b(pcT, NCH, Nt * 16, ks * 32);
        gacc = wmma16(a, bb, gacc);
      }
    }
  }
  __syncthreads();
  if (w < 4) {  // store G transposed: GB[t2][t1], padding stays zero
#pragma unroll
    for (int r = 0; r < 8; r++) {
      int t1 = Mt * 16 + r + 8 * hi;
      int t2 = Nt * 16 + col;
      if (t1 < Tt && t2 < Tt) GB[t2 * 32 + t1] = (_Float16)gacc[r];
    }
  }
  __syncthreads();

  // stage ft rows (512 x 32, zero-padded t' >= 24), reusing S
  for (int i = tid; i < 512 * 32; i += 256) {
    int m = i >> 5, t = i & 31;
    _Float16 v = (_Float16)0.f;
    if (t < Tt) v = FT[(((size_t)(b * Mm + m)) * CHN + c) * Tt + t];
    S[i] = v;
  }
  __syncthreads();

  // pt[m, t] = ft[m, :] @ G ; 8 waves x 4 row-blocks x 2 n-tiles
#pragma unroll
  for (int blk = 0; blk < 4; blk++) {
    int m0 = (w * 4 + blk) * 16;
    v16h a = load_frag_a(S, 32, m0, 0);
#pragma unroll
    for (int nt = 0; nt < 2; nt++) {
      v8f z = {0.f, 0.f, 0.f, 0.f, 0.f, 0.f, 0.f, 0.f};
      z = wmma16(a, load_frag_b(GB, 32, nt * 16, 0), z);
      int t = nt * 16 + col;
      if (t < Tt) {
#pragma unroll
        for (int r = 0; r < 8; r++) {
          int m = m0 + r + 8 * hi;
          PTpm[(((size_t)(b * Mm + m)) * Tt + t) * (size_t)CHN + c] = z[r];
        }
      }
    }
  }
}

// ---------------------------------------------------------------------------
// LayerNorm over channels (fused into staging) + 4-layer 64->64 MLP, f32 out.
// Input is point-major [point][64] f32 so each point's row is contiguous.
// ---------------------------------------------------------------------------
__global__ void __launch_bounds__(256)
mlp4_out_kernel(const float* __restrict__ PT, float* __restrict__ Out,
                const float* __restrict__ W0, const float* __restrict__ Bi0,
                const float* __restrict__ W1, const float* __restrict__ Bi1,
                const float* __restrict__ W2, const float* __restrict__ Bi2,
                const float* __restrict__ W3, const float* __restrict__ Bi3,
                const float* __restrict__ gamma, const float* __restrict__ beta,
                int P) {
  __shared__ __align__(128) _Float16 Xs[128 * 72];
  __shared__ __align__(128) _Float16 Ws[4 * 64 * 64];
  __shared__ float Bs[4 * 64];

  const int tid = (int)threadIdx.x;
  for (int i = tid; i < 64 * 64; i += 256) {
    Ws[i]         = (_Float16)W0[i];
    Ws[4096 + i]  = (_Float16)W1[i];
    Ws[8192 + i]  = (_Float16)W2[i];
    Ws[12288 + i] = (_Float16)W3[i];
  }
  if (tid < 64) {
    Bs[tid]       = Bi0[tid];
    Bs[64 + tid]  = Bi1[tid];
    Bs[128 + tid] = Bi2[tid];
    Bs[192 + tid] = Bi3[tid];
  }

  // LayerNorm over the contiguous 64-channel row, fused into f16 staging
  const int p0 = (int)blockIdx.x * 128;
  if (tid < 128) {
    int p = p0 + tid;
    if (p < P) {
      const float* row = PT + (size_t)p * 64;
      float s = 0.f, ss = 0.f;
      for (int cc = 0; cc < 64; cc++) { float v = row[cc]; s += v; ss += v * v; }
      float mu = s * (1.f / 64.f);
      float var = ss * (1.f / 64.f) - mu * mu;
      float rstd = rsqrtf(var + LN_EPS);
      for (int cc = 0; cc < 64; cc++)
        Xs[tid * 72 + cc] = (_Float16)((row[cc] - mu) * rstd * gamma[cc] + beta[cc]);
    } else {
      for (int cc = 0; cc < 64; cc++) Xs[tid * 72 + cc] = (_Float16)0.f;
    }
  }
  __syncthreads();

  const int w = tid >> 5, m0 = w * 16;
  const int l = lane_id(), col = l & 15, hi = l >> 4;

#pragma unroll
  for (int L = 0; L < 4; L++) {
    const _Float16* Wl = Ws + L * 4096;
    v16h a0 = load_frag_a(Xs, 72, m0, 0);
    v16h a1 = load_frag_a(Xs, 72, m0, 32);

    v8f acc[4];
#pragma unroll
    for (int nt = 0; nt < 4; nt++) {
      v8f z = {0.f, 0.f, 0.f, 0.f, 0.f, 0.f, 0.f, 0.f};
      z = wmma16(a0, load_frag_b(Wl, 64, nt * 16, 0), z);
      z = wmma16(a1, load_frag_b(Wl, 64, nt * 16, 32), z);
      acc[nt] = z;
    }

    if (L < 3) {
#pragma unroll
      for (int nt = 0; nt < 4; nt++) {
        int o = nt * 16 + col;
        float bia = Bs[L * 64 + o];
#pragma unroll
        for (int r = 0; r < 8; r++) {
          int m = m0 + r + 8 * hi;
          float v = acc[nt][r] + bia;
          Xs[m * 72 + o] = (_Float16)(v > 0.f ? v : 0.f);
        }
      }
    } else {
#pragma unroll
      for (int nt = 0; nt < 4; nt++) {
        int o = nt * 16 + col;
        float bia = Bs[192 + o];
#pragma unroll
        for (int r = 0; r < 8; r++) {
          int m = m0 + r + 8 * hi;
          int p = p0 + m;
          if (p < P) {
            int row = p / Tt, t = p - row * Tt;
            Out[((size_t)row * CHN + o) * Tt + t] = acc[nt][r] + bia;  // (b,m,o,t)
          }
        }
      }
    }
  }
}

// ---------------------------------------------------------------------------
extern "C" void kernel_launch(void* const* d_in, const int* in_sizes, int n_in,
                              void* d_out, int out_size, void* d_ws, size_t ws_size,
                              hipStream_t stream) {
  (void)in_sizes; (void)n_in; (void)out_size; (void)ws_size;

  const float* feat_target  = (const float*)d_in[0];
  const float* feat_context = (const float*)d_in[1];
  const float* pred_context = (const float*)d_in[2];
  // feat_params: (W,b) x4 -> indices 3..10
  const float* fW0 = (const float*)d_in[3];  const float* fB0 = (const float*)d_in[4];
  const float* fW1 = (const float*)d_in[5];  const float* fB1 = (const float*)d_in[6];
  const float* fW2 = (const float*)d_in[7];  const float* fB2 = (const float*)d_in[8];
  const float* fW3 = (const float*)d_in[9];  const float* fB3 = (const float*)d_in[10];
  // pred_params: 11..18
  const float* pW0 = (const float*)d_in[11]; const float* pB0 = (const float*)d_in[12];
  const float* pW1 = (const float*)d_in[13]; const float* pB1 = (const float*)d_in[14];
  const float* pW2 = (const float*)d_in[15]; const float* pB2 = (const float*)d_in[16];
  const float* pW3 = (const float*)d_in[17]; const float* pB3 = (const float*)d_in[18];
  // out_params: 19..26
  const float* oW0 = (const float*)d_in[19]; const float* oB0 = (const float*)d_in[20];
  const float* oW1 = (const float*)d_in[21]; const float* oB1 = (const float*)d_in[22];
  const float* oW2 = (const float*)d_in[23]; const float* oB2 = (const float*)d_in[24];
  const float* oW3 = (const float*)d_in[25]; const float* oB3 = (const float*)d_in[26];
  const float* ln_gamma = (const float*)d_in[27];
  const float* ln_beta  = (const float*)d_in[28];

  // workspace: ft, fc, pc as f16 (6 MB each), pt point-major f32 (12.6 MB)
  const size_t halfElems = (size_t)Bb * Mm * CHN * Tt;  // 3,145,728
  _Float16* ft = (_Float16*)d_ws;
  _Float16* fc = ft + halfElems;
  _Float16* pc = fc + halfElems;
  float*    pt = (float*)(pc + halfElems);

  const int P = Bb * Mm * Tt;  // 49152 points
  dim3 blk(256);
  int gridP = (P + 127) / 128;  // 384

  mlp4_small_kernel<<<gridP, blk, 0, stream>>>(feat_target, ft,
      fW0, fB0, fW1, fB1, fW2, fB2, fW3, fB3, /*cin=*/7, P);
  mlp4_small_kernel<<<gridP, blk, 0, stream>>>(feat_context, fc,
      fW0, fB0, fW1, fB1, fW2, fB2, fW3, fB3, /*cin=*/7, P);
  mlp4_small_kernel<<<gridP, blk, 0, stream>>>(pred_context, pc,
      pW0, pB0, pW1, pB1, pW2, pB2, pW3, pB3, /*cin=*/1, P);

  adj_apply_kernel<<<Bb * CHN, blk, 0, stream>>>(ft, fc, pc, pt);

  mlp4_out_kernel<<<gridP, blk, 0, stream>>>(pt, (float*)d_out,
      oW0, oB0, oW1, oB1, oW2, oB2, oW3, oB3, ln_gamma, ln_beta, P);
}
